// MultiHeadAttentionBlock_25924422598728
// MI455X (gfx1250) — matrix operand order repaired and checked by reading, not compiled, not run on hardware
//
#include <hip/hip_runtime.h>

#define S_LEN  2048
#define NB     2
#define NH     16
#define DKK    64
#define DMODEL 1024
#define NROWS  (NB * S_LEN)   // 4096

typedef __bf16 bf16_t;
typedef __attribute__((ext_vector_type(16))) __bf16 v16bf;
typedef __attribute__((ext_vector_type(8)))  __bf16 v8bf;
typedef __attribute__((ext_vector_type(8)))  float  v8f;

// ---------------------------------------------------------------------------
// WMMA fragment loaders (wave32, 16x16x32 bf16).
//
// A fragment (16x32, M x K). Lane half hx = lane>>4, row = lane&15:
//   elems 0..7  <- K = k0 + hx*16 + i       (VGPR 0..3, 2 bf16/VGPR)
//   elems 8..15 <- K = k0 + hx*16 + 8 + i   (VGPR 4..7)
static __device__ __forceinline__ v16bf afrag(const bf16_t* row, int k0, int hx) {
  v8bf lo = *reinterpret_cast<const v8bf*>(row + k0 + hx * 16);
  v8bf hi = *reinterpret_cast<const v8bf*>(row + k0 + hx * 16 + 8);
  v16bf r;
#pragma unroll
  for (int i = 0; i < 8; ++i) { r[i] = lo[i]; r[i + 8] = hi[i]; }
  return r;
}
static __device__ __forceinline__ v16bf afrag(const float* row, int k0, int hx) {
  v8f lo = *reinterpret_cast<const v8f*>(row + k0 + hx * 16);
  v8f hi = *reinterpret_cast<const v8f*>(row + k0 + hx * 16 + 8);
  v16bf r;
#pragma unroll
  for (int i = 0; i < 8; ++i) { r[i] = (bf16_t)lo[i]; r[i + 8] = (bf16_t)hi[i]; }
  return r;
}
// B fragment (32x16, K x N). Lane = one column n = lane&15, contiguous K:
//   elems 0..15 <- K = k0 + hx*16 + i   (one 32-byte load)
static __device__ __forceinline__ v16bf bfrag(const bf16_t* row, int k0, int hx) {
  return *reinterpret_cast<const v16bf*>(row + k0 + hx * 16);
}
static __device__ __forceinline__ v16bf bfrag(const float* row, int k0, int hx) {
  v8f lo = *reinterpret_cast<const v8f*>(row + k0 + hx * 16);
  v8f hi = *reinterpret_cast<const v8f*>(row + k0 + hx * 16 + 8);
  v16bf r;
#pragma unroll
  for (int i = 0; i < 8; ++i) { r[i] = (bf16_t)lo[i]; r[i + 8] = (bf16_t)hi[i]; }
  return r;
}

static __device__ __forceinline__ v8f wmma_bf(v16bf a, v16bf b, v8f c) {
  return __builtin_amdgcn_wmma_f32_16x16x32_bf16(false, a, false, b,
                                                 (short)0, c, false, false);
}

// ---------------------------------------------------------------------------
// GEMM: C[M=4096, N=1024] = A[4096,1024] * W[1024,1024]^T  (y = x @ W.T)
// One wave = 16x64 strip of C (A-frag reused over 4 N-subtiles).
// mode 0: f32 row-major to out   (final projection -> d_out)
// mode 1: bf16 head-major [B,H,S,dk]           (Q, K)
// mode 2: bf16 head-major transposed [B,H,dk,S] (V, for contiguous-K B frags)
template <typename AT>
__global__ void __launch_bounds__(128)
gemm_bf16_wmma(const AT* __restrict__ A, const float* __restrict__ W,
               void* __restrict__ outp, const int mode) {
  const int lane = threadIdx.x & 31;
  const int wv   = threadIdx.x >> 5;
  const int tile = blockIdx.x * 4 + wv;
  const int nt64 = DMODEL / 64;                 // 16 column strips
  const int m0 = (tile / nt64) * 16;
  const int n0 = (tile % nt64) * 64;
  const int hx = lane >> 4, r = lane & 15;

  v8f acc[4] = {};
  const AT* arow = A + (size_t)(m0 + r) * DMODEL;
  for (int k0 = 0; k0 < DMODEL; k0 += 32) {
    v16bf a = afrag(arow, k0, hx);
#pragma unroll
    for (int t = 0; t < 4; ++t) {
      v16bf bm = bfrag(W + (size_t)(n0 + t * 16 + r) * DMODEL, k0, hx);
      acc[t] = wmma_bf(a, bm, acc[t]);
    }
  }

  // C layout: VGPR g holds row (hx*8+g), column (lane&15) of the 16x16 tile.
  if (mode == 0) {
    float* O = reinterpret_cast<float*>(outp);
#pragma unroll
    for (int t = 0; t < 4; ++t)
#pragma unroll
      for (int g = 0; g < 8; ++g)
        O[(size_t)(m0 + hx * 8 + g) * DMODEL + n0 + t * 16 + r] = acc[t][g];
  } else if (mode == 1) {
    bf16_t* O = reinterpret_cast<bf16_t*>(outp);
#pragma unroll
    for (int t = 0; t < 4; ++t)
#pragma unroll
      for (int g = 0; g < 8; ++g) {
        int m = m0 + hx * 8 + g, n = n0 + t * 16 + r;
        int b = m >> 11, s = m & (S_LEN - 1), h = n >> 6, d = n & (DKK - 1);
        O[(((size_t)(b * NH + h)) * S_LEN + s) * DKK + d] = (bf16_t)acc[t][g];
      }
  } else {
    bf16_t* O = reinterpret_cast<bf16_t*>(outp);
#pragma unroll
    for (int t = 0; t < 4; ++t)
#pragma unroll
      for (int g = 0; g < 8; ++g) {
        int m = m0 + hx * 8 + g, n = n0 + t * 16 + r;
        int b = m >> 11, s = m & (S_LEN - 1), h = n >> 6, d = n & (DKK - 1);
        O[(((size_t)(b * NH + h)) * DKK + d) * S_LEN + s] = (bf16_t)acc[t][g];
      }
  }
}

// ---------------------------------------------------------------------------
// Flash attention: one wave owns a 16-row Q tile of one (b,h); streams K/V in
// j-chunks of 32 with online softmax. Scores never hit HBM.
__global__ void __launch_bounds__(128)
attn_fa_kernel(const bf16_t* __restrict__ Qh, const bf16_t* __restrict__ Kh,
               const bf16_t* __restrict__ Vt, const int* __restrict__ mask,
               bf16_t* __restrict__ ctx) {
  __shared__ bf16_t pbuf[4][16][32];            // per-wave P staging (C->A relayout)
  const int lane = threadIdx.x & 31;
  const int wv   = threadIdx.x >> 5;
  const int tile = blockIdx.x * 4 + wv;
  const int qt = tile & (S_LEN / 16 - 1);       // 128 q-tiles per head
  const int bh = tile >> 7;
  const int b  = bh >> 4;
  const int h  = bh & (NH - 1);
  const int q0 = qt * 16;
  const int hx = lane >> 4, r = lane & 15;

  const bf16_t* Qrow = Qh + ((size_t)bh * S_LEN + q0 + r) * DKK;
  const v16bf aQ0 = afrag(Qrow, 0, hx);         // d = 0..31
  const v16bf aQ1 = afrag(Qrow, 32, hx);        // d = 32..63
  const bf16_t* Kbh = Kh + (size_t)bh * S_LEN * DKK;
  const bf16_t* Vbh = Vt + (size_t)bh * DKK * S_LEN;
  const int* Mbase = mask + ((size_t)b * S_LEN + q0) * S_LEN;

  float mrun[8], lrun[8];
  v8f acc[4] = {};
#pragma unroll
  for (int g = 0; g < 8; ++g) { mrun[g] = -1e30f; lrun[g] = 0.0f; }

  for (int j0 = 0; j0 < S_LEN; j0 += 32) {
    // S = Q * K^T for 32 j-columns: two 16x16 tiles, K-depth 64 = 2 wmma each.
    v8f s0 = {}, s1 = {};
    {
      const bf16_t* k0r = Kbh + (size_t)(j0 + r) * DKK;
      const bf16_t* k1r = Kbh + (size_t)(j0 + 16 + r) * DKK;
      s0 = wmma_bf(aQ0, bfrag(k0r, 0, hx), s0);
      s0 = wmma_bf(aQ1, bfrag(k0r, 32, hx), s0);
      s1 = wmma_bf(aQ0, bfrag(k1r, 0, hx), s1);
      s1 = wmma_bf(aQ1, bfrag(k1r, 32, hx), s1);
    }
    // mask + scale + online softmax (row stats reduced across 16-lane half).
#pragma unroll
    for (int g = 0; g < 8; ++g) {
      const int row = hx * 8 + g;
      const int* mr = Mbase + (size_t)row * S_LEN + j0;
      float x0 = s0[g] * 0.125f + (mr[r]      == 0 ? -1e9f : 0.0f);
      float x1 = s1[g] * 0.125f + (mr[16 + r] == 0 ? -1e9f : 0.0f);
      float mx = fmaxf(x0, x1);
      mx = fmaxf(mx, __shfl_xor(mx, 1, 32));
      mx = fmaxf(mx, __shfl_xor(mx, 2, 32));
      mx = fmaxf(mx, __shfl_xor(mx, 4, 32));
      mx = fmaxf(mx, __shfl_xor(mx, 8, 32));
      float mnew  = fmaxf(mrun[g], mx);
      float alpha = __expf(mrun[g] - mnew);
      float p0 = __expf(x0 - mnew);
      float p1 = __expf(x1 - mnew);
      float rs = p0 + p1;
      rs += __shfl_xor(rs, 1, 32);
      rs += __shfl_xor(rs, 2, 32);
      rs += __shfl_xor(rs, 4, 32);
      rs += __shfl_xor(rs, 8, 32);
      lrun[g] = lrun[g] * alpha + rs;
      mrun[g] = mnew;
#pragma unroll
      for (int t = 0; t < 4; ++t) acc[t][g] = acc[t][g] * alpha;
      pbuf[wv][row][r]      = (bf16_t)p0;       // C layout -> row-major LDS
      pbuf[wv][row][16 + r] = (bf16_t)p1;
    }
    // Wave-local LDS fence (CDNA5 split counter), then reload P as A fragment.
    asm volatile("s_wait_dscnt 0" ::: "memory");
    v16bf aP = afrag(&pbuf[wv][r][0], 0, hx);
    // O += P * V  (V^T rows give contiguous-K B fragments).
#pragma unroll
    for (int t = 0; t < 4; ++t) {
      const bf16_t* vrow = Vbh + (size_t)(t * 16 + r) * S_LEN;
      acc[t] = wmma_bf(aP, bfrag(vrow, j0, hx), acc[t]);
    }
  }

  // Normalize and write context row-major [4096, 1024] (heads re-merged).
#pragma unroll
  for (int g = 0; g < 8; ++g) {
    const int row = hx * 8 + g;
    const float inv = 1.0f / lrun[g];
    bf16_t* crow = ctx + ((size_t)b * S_LEN + q0 + row) * DMODEL + h * DKK;
#pragma unroll
    for (int t = 0; t < 4; ++t) crow[t * 16 + r] = (bf16_t)(acc[t][g] * inv);
  }
}

// ---------------------------------------------------------------------------
extern "C" void kernel_launch(void* const* d_in, const int* in_sizes, int n_in,
                              void* d_out, int out_size, void* d_ws, size_t ws_size,
                              hipStream_t stream) {
  (void)in_sizes; (void)n_in; (void)out_size; (void)ws_size;
  const float* q    = (const float*)d_in[0];
  const float* k    = (const float*)d_in[1];
  const float* v    = (const float*)d_in[2];
  const int*   mask = (const int*)d_in[3];
  const float* w_q  = (const float*)d_in[4];
  const float* w_k  = (const float*)d_in[5];
  const float* w_v  = (const float*)d_in[6];
  const float* w_o  = (const float*)d_in[7];

  // Workspace: 4 x (4096*1024) bf16 = 32 MiB.
  const size_t nelem = (size_t)NROWS * DMODEL;
  bf16_t* Qh = (bf16_t*)d_ws;   // [B,H,S,dk]
  bf16_t* Kh = Qh + nelem;      // [B,H,S,dk]
  bf16_t* Vt = Kh + nelem;      // [B,H,dk,S]
  bf16_t* cx = Vt + nelem;      // [B*S, D] context

  dim3 blk(128);
  dim3 gg((NROWS / 16) * (DMODEL / 64) / 4);    // 1024 blocks, 4 waves each
  gemm_bf16_wmma<float><<<gg, blk, 0, stream>>>(q, w_q, (void*)Qh, 1);
  gemm_bf16_wmma<float><<<gg, blk, 0, stream>>>(k, w_k, (void*)Kh, 1);
  gemm_bf16_wmma<float><<<gg, blk, 0, stream>>>(v, w_v, (void*)Vt, 2);

  dim3 ga((NB * NH * (S_LEN / 16)) / 4);        // 1024 blocks
  attn_fa_kernel<<<ga, blk, 0, stream>>>(Qh, Kh, Vt, mask, cx);

  gemm_bf16_wmma<bf16_t><<<gg, blk, 0, stream>>>(cx, w_o, d_out, 0);
}
